// FTPPIOneHot_86517821216492
// MI455X (gfx1250) — matrix operand order repaired and checked by reading, not compiled, not run on hardware
//
#include <hip/hip_runtime.h>
#include <hip/hip_bf16.h>
#include <math.h>

// ---------------------------------------------------------------------------
// Problem constants (match reference)
// ---------------------------------------------------------------------------
#define VOCAB   32000
#define HIDDEN  1024
#define BATCH   128
#define SEQ     2048
#define NEG_INF (-1.0e9f)

// Split-K GEMM config: K = 32000 = 50 chunks * 640.
// Block computes M=128 x N=32 per chunk; each round stages K=64 x N=32 of w1
// in LDS -> 2 k-steps x 2 n-tiles = 4 wmma per barrier pair.
#define KSPLIT  50
#define KCHUNK  (VOCAB / KSPLIT)   // 640
#define KDEPTH  64                 // K staged in LDS per barrier round
#define KROUNDS (KCHUNK / KDEPTH)  // 10
#define KSUB    (KDEPTH / 32)      // 2 wmma k-steps per round
#define NTILES  2                  // 16-wide n-tiles per block (N=32)

typedef __attribute__((ext_vector_type(16))) __bf16 v16bf;
typedef __attribute__((ext_vector_type(8)))  float  v8f;

// Native f32 -> bf16 conversion (clang lowers to hardware cvt, RTNE).
static __device__ __forceinline__ __bf16 f2bf(float f) {
  return (__bf16)f;
}

// ---------------------------------------------------------------------------
// Kernel 0: zero pooled[128*32000] and h[128*1024] (contiguous in ws)
// ---------------------------------------------------------------------------
__global__ void ftpp_zero_kernel(float* __restrict__ p, int n) {
  int i = blockIdx.x * blockDim.x + threadIdx.x;
  int stride = gridDim.x * blockDim.x;
  for (; i < n; i += stride) p[i] = 0.0f;
}

// ---------------------------------------------------------------------------
// Kernel 1: per-batch masked softmax over SEQ + scatter-add into pooled[b][v]
// One block (256 threads) per batch row; tokens/logits cached in registers.
// ---------------------------------------------------------------------------
__global__ void ftpp_softmax_scatter_kernel(const int* __restrict__ x,
                                            const float* __restrict__ w_attn,
                                            const float* __restrict__ b_attn,
                                            float* __restrict__ pooled) {
  __shared__ float red[256];
  const int b   = blockIdx.x;
  const int tid = threadIdx.x;
  const float bias = b_attn[0];

  int   xs[SEQ / 256];
  float ls[SEQ / 256];
  float lmax = -3.0e38f;
#pragma unroll
  for (int j = 0; j < SEQ / 256; ++j) {
    int s  = tid + j * 256;
    int xi = x[b * SEQ + s];
    float l = (xi > 0) ? (w_attn[xi] + bias) : NEG_INF;
    xs[j] = xi;
    ls[j] = l;
    lmax  = fmaxf(lmax, l);
  }
  red[tid] = lmax;
  __syncthreads();
  for (int off = 128; off > 0; off >>= 1) {
    if (tid < off) red[tid] = fmaxf(red[tid], red[tid + off]);
    __syncthreads();
  }
  lmax = red[0];
  __syncthreads();

  float sum = 0.0f;
#pragma unroll
  for (int j = 0; j < SEQ / 256; ++j) {
    float e = __expf(ls[j] - lmax);
    ls[j] = e;
    sum  += e;
  }
  red[tid] = sum;
  __syncthreads();
  for (int off = 128; off > 0; off >>= 1) {
    if (tid < off) red[tid] += red[tid + off];
    __syncthreads();
  }
  const float inv = 1.0f / red[0];

#pragma unroll
  for (int j = 0; j < SEQ / 256; ++j) {
    atomicAdd(&pooled[(size_t)b * VOCAB + xs[j]], ls[j] * inv);
  }
}

// ---------------------------------------------------------------------------
// Kernel 2: pooled f32 -> bf16 (A matrix for WMMA, row-major [128][32000])
// ---------------------------------------------------------------------------
__global__ void ftpp_convert_kernel(const float* __restrict__ src,
                                    __bf16* __restrict__ dst, int n) {
  int i = blockIdx.x * blockDim.x + threadIdx.x;
  int stride = gridDim.x * blockDim.x;
  for (; i < n; i += stride) dst[i] = f2bf(src[i]);
}

// ---------------------------------------------------------------------------
// Kernel 3: split-K bf16 WMMA GEMM: h[128][1024] += A[128][32000] * w1[32000][1024]
// Block = 256 thr = 8 waves; block owns all 8 M-strips for a 32-wide N tile
// over one 640-deep K chunk (w1 loaded exactly once per chunk). Each wave
// reuses one A fragment for two WMMAs (two n-tiles) -> halved A traffic.
// B staged through LDS pre-swizzled into per-(step,ntile) fragment order;
// next round's w1 lines prefetched (global_prefetch_b8) during WMMA work.
// ---------------------------------------------------------------------------
__global__ void ftpp_gemm_kernel(const __bf16* __restrict__ A,   // bf16 [128][32000]
                                 const float* __restrict__ w1,   // f32  [32000][1024]
                                 float* __restrict__ h) {        // f32  [128][1024]
  __shared__ __bf16 bl[KSUB * NTILES * 512];  // 4 x (32x16) bf16 frag tiles

  const int tid   = threadIdx.x;
  const int lane  = tid & 31;
  const int wv    = tid >> 5;            // 0..7 -> M strip [16*wv, 16*wv+16)
  const int n0    = blockIdx.x * (16 * NTILES);
  const int k0    = blockIdx.y * KCHUNK;
  const int m     = (wv << 4) + (lane & 15);
  const int khalf = (lane >> 4) << 3;    // 0 for lanes 0-15, 8 for lanes 16-31

  // Staging decomposition: thread handles 8 consecutive floats of one w1 row.
  const int t8   = tid * 8;
  const int kkL  = t8 >> 5;              // local k row 0..63 (KDEPTH rows)
  const int nn0  = t8 & 31;              // 0,8,16,24 within the 32-wide N tile
  const int ntS  = nn0 >> 4;             // n-tile 0/1
  const int nfS  = nn0 & 15;             // 0 or 8 within fragment
  const int sS   = kkL >> 5;             // wmma k-step 0/1
  const int klS  = kkL & 31;
  const int iS   = (klS & 7) | ((klS & 16) >> 1);  // fragment element index
  const int lbS  = (klS & 8) << 1;       // fragment lane base (0 or 16)
  __bf16* stg = &bl[((((sS << 1) | ntS)) << 9) | iS];

  const __bf16* aRow = &A[(size_t)m * VOCAB + k0 + khalf];
  const float* srcBase = &w1[(size_t)(k0 + kkL) * HIDDEN + n0 + nn0];

  v8f c0 = {0.f, 0.f, 0.f, 0.f, 0.f, 0.f, 0.f, 0.f};
  v8f c1 = {0.f, 0.f, 0.f, 0.f, 0.f, 0.f, 0.f, 0.f};

  for (int round = 0; round < KROUNDS; ++round) {
    // Cooperative load: KDEPTH x 32 f32 of w1 -> bf16 -> LDS (fragment order).
    {
      const float* src = srcBase + (size_t)round * KDEPTH * HIDDEN;
      float4 v0 = *(const float4*)(src);
      float4 v1 = *(const float4*)(src + 4);
      if (round + 1 < KROUNDS) {
        // Pull next round's lines toward L2/L0 while this round's WMMAs run.
        __builtin_prefetch(src + (size_t)KDEPTH * HIDDEN, 0, 1);
      }
      stg[(lbS | (nfS + 0)) << 4] = f2bf(v0.x);
      stg[(lbS | (nfS + 1)) << 4] = f2bf(v0.y);
      stg[(lbS | (nfS + 2)) << 4] = f2bf(v0.z);
      stg[(lbS | (nfS + 3)) << 4] = f2bf(v0.w);
      stg[(lbS | (nfS + 4)) << 4] = f2bf(v1.x);
      stg[(lbS | (nfS + 5)) << 4] = f2bf(v1.y);
      stg[(lbS | (nfS + 6)) << 4] = f2bf(v1.z);
      stg[(lbS | (nfS + 7)) << 4] = f2bf(v1.w);
    }
    __syncthreads();

#pragma unroll
    for (int s = 0; s < KSUB; ++s) {
      // A fragment: 16-bit A layout 16x32 — two contiguous 16B global loads.
      // lanes 0-15: elems 0-7 = K[0..7], 8-15 = K[16..23]; lanes 16-31: +8.
      v16bf a, bf0, bf1;
      const __bf16* ap = aRow + round * KDEPTH + s * 32;
      ((uint4*)&a)[0] = *(const uint4*)(ap);
      ((uint4*)&a)[1] = *(const uint4*)(ap + 16);

      // B fragments: 32B contiguous per lane from pre-swizzled LDS.
      const __bf16* bp0 = &bl[((s << 1) << 9) | (lane << 4)];
      const __bf16* bp1 = &bl[(((s << 1) | 1) << 9) | (lane << 4)];
      ((uint4*)&bf0)[0] = ((const uint4*)bp0)[0];
      ((uint4*)&bf0)[1] = ((const uint4*)bp0)[1];
      ((uint4*)&bf1)[0] = ((const uint4*)bp1)[0];
      ((uint4*)&bf1)[1] = ((const uint4*)bp1)[1];

      c0 = __builtin_amdgcn_wmma_f32_16x16x32_bf16(
          false, a, false, bf0, (short)0, c0, false, false);
      c1 = __builtin_amdgcn_wmma_f32_16x16x32_bf16(
          false, a, false, bf1, (short)0, c1, false, false);
    }
    __syncthreads();
  }

  // C/D layout: VGPR j -> M = j + 8*(lane>=16), N = lane & 15. Split-K: atomic.
  const int col   = lane & 15;
  const int rbase = (lane >> 4) << 3;
#pragma unroll
  for (int j = 0; j < 8; ++j) {
    const int row = (wv << 4) + rbase + j;
    atomicAdd(&h[(size_t)row * HIDDEN + n0 + col],      c0[j]);
    atomicAdd(&h[(size_t)row * HIDDEN + n0 + 16 + col], c1[j]);
  }
}

// ---------------------------------------------------------------------------
// Kernel 4: head: relu(h + b1) @ w2 + b2 -> log_softmax, one block per batch
// ---------------------------------------------------------------------------
__global__ void ftpp_head_kernel(const float* __restrict__ h,
                                 const float* __restrict__ b1,
                                 const float* __restrict__ w2,
                                 const float* __restrict__ b2,
                                 float* __restrict__ out) {
  __shared__ float r0[256];
  __shared__ float r1[256];
  const int b   = blockIdx.x;
  const int tid = threadIdx.x;

  float p0 = 0.0f, p1 = 0.0f;
  for (int i = tid; i < HIDDEN; i += 256) {
    float v = h[(size_t)b * HIDDEN + i] + b1[i];
    v = fmaxf(v, 0.0f);
    p0 += v * w2[i * 2 + 0];
    p1 += v * w2[i * 2 + 1];
  }
  r0[tid] = p0;
  r1[tid] = p1;
  __syncthreads();
  for (int off = 128; off > 0; off >>= 1) {
    if (tid < off) { r0[tid] += r0[tid + off]; r1[tid] += r1[tid + off]; }
    __syncthreads();
  }
  if (tid == 0) {
    float l0 = r0[0] + b2[0];
    float l1 = r1[0] + b2[1];
    float mx  = fmaxf(l0, l1);
    float lse = mx + logf(__expf(l0 - mx) + __expf(l1 - mx));
    out[b * 2 + 0] = l0 - lse;
    out[b * 2 + 1] = l1 - lse;
  }
}

// ---------------------------------------------------------------------------
// Launch
// ---------------------------------------------------------------------------
extern "C" void kernel_launch(void* const* d_in, const int* in_sizes, int n_in,
                              void* d_out, int out_size, void* d_ws, size_t ws_size,
                              hipStream_t stream) {
  const int*   x      = (const int*)  d_in[0];
  const float* w_attn = (const float*)d_in[1];
  const float* b_attn = (const float*)d_in[2];
  const float* w1     = (const float*)d_in[3];
  const float* b1     = (const float*)d_in[4];
  const float* w2     = (const float*)d_in[5];
  const float* b2     = (const float*)d_in[6];
  float* out = (float*)d_out;

  // Workspace layout (25.1 MB total):
  //   pooled f32 [128*32000]  @ 0          (16,384,000 B)
  //   h      f32 [128*1024]   @ 16,384,000 (   524,288 B)   <- contiguous w/ pooled
  //   pooled bf16 [128*32000] @ 16,908,288 ( 8,192,000 B)
  char* ws = (char*)d_ws;
  float*  pooled   = (float*)ws;
  float*  h        = (float*)(ws + 16384000);
  __bf16* pooledBf = (__bf16*)(ws + 16384000 + 524288);

  const int nzero = BATCH * VOCAB + BATCH * HIDDEN;  // pooled + h, contiguous
  ftpp_zero_kernel<<<2048, 256, 0, stream>>>(pooled, nzero);
  ftpp_softmax_scatter_kernel<<<BATCH, 256, 0, stream>>>(x, w_attn, b_attn, pooled);
  ftpp_convert_kernel<<<2048, 256, 0, stream>>>(pooled, pooledBf, BATCH * VOCAB);
  ftpp_gemm_kernel<<<dim3(HIDDEN / (16 * NTILES), KSPLIT), 256, 0, stream>>>(pooledBf, w1, h);
  ftpp_head_kernel<<<BATCH, 256, 0, stream>>>(h, b1, w2, b2, out);
}
